// AdvancedTransformerBlock_72258529788252
// MI455X (gfx1250) — matrix operand order repaired and checked by reading, not compiled, not run on hardware
//
#include <hip/hip_runtime.h>
#include <hip/hip_bf16.h>
#include <math.h>
#include <stdint.h>

typedef __attribute__((ext_vector_type(16))) __bf16 v16bf;
typedef __attribute__((ext_vector_type(8)))  float  v8f;

// Problem constants (match reference)
static constexpr int Bc   = 2;
static constexpr int Sc   = 2048;
static constexpr int Dc   = 2048;
static constexpr int Hc   = 16;
static constexpr int DHc  = 128;
static constexpr int HFFc = 8192;
static constexpr int BSc  = Bc * Sc;            // 4096 rows

// ---------------------------------------------------------------------------
// CDNA5 async global->LDS copy (ASYNCcnt-tracked), 16 bytes per lane.
// lds_addr is the 32-bit LDS byte offset (low 32 bits of generic address).
// ---------------------------------------------------------------------------
__device__ __forceinline__ void async_copy_b128(uint32_t lds_addr, const void* gptr) {
    asm volatile("global_load_async_to_lds_b128 %0, %1, off"
                 :
                 : "v"(lds_addr), "v"((uint64_t)(uintptr_t)gptr)
                 : "memory");
}
__device__ __forceinline__ void async_wait0() {
    asm volatile("s_wait_asynccnt 0" ::: "memory");
}
__device__ __forceinline__ uint32_t lds_off(const void* p) {
    return (uint32_t)(uintptr_t)p;      // LDS aperture: addr[31:0] == LDS offset
}

// ---------------------------------------------------------------------------
// fp32 -> bf16 cast (grid-stride)
// ---------------------------------------------------------------------------
__global__ void cast_f32_bf16_kernel(const float* __restrict__ in,
                                     __bf16* __restrict__ out, size_t n) {
    size_t i = (size_t)blockIdx.x * blockDim.x + threadIdx.x;
    size_t stride = (size_t)gridDim.x * blockDim.x;
    for (; i < n; i += stride) out[i] = (__bf16)in[i];
}

// ---------------------------------------------------------------------------
// RMSNorm: out_bf16[row, :] = w * x / sqrt(mean(x^2) + eps)
// ---------------------------------------------------------------------------
__global__ __launch_bounds__(256)
void rmsnorm_kernel(const float* __restrict__ x, const float* __restrict__ w,
                    __bf16* __restrict__ out, int ncols) {
    const int row = blockIdx.x;
    const float* xr = x + (size_t)row * ncols;
    float ss = 0.f;
    for (int c = threadIdx.x; c < ncols; c += blockDim.x) {
        float v = xr[c];
        ss += v * v;
    }
    #pragma unroll
    for (int m = 16; m >= 1; m >>= 1) ss += __shfl_xor(ss, m, 32);
    __shared__ float wsum[8];
    const int lane = threadIdx.x & 31, wv = threadIdx.x >> 5;
    if (lane == 0) wsum[wv] = ss;
    __syncthreads();
    float tot = 0.f;
    #pragma unroll
    for (int i = 0; i < 8; i++) tot += wsum[i];
    const float rinv = rsqrtf(tot / (float)ncols + 1e-5f);
    __bf16* orow = out + (size_t)row * ncols;
    for (int c = threadIdx.x; c < ncols; c += blockDim.x)
        orow[c] = (__bf16)(w[c] * xr[c] * rinv);
}

// ---------------------------------------------------------------------------
// Tiled bf16 WMMA GEMM:  C[M,N] = A[M,K] @ W[N,K]^T
// 128x128 block tile, BK=32, 256 threads = 8 waves (4 x 2 wave grid).
// Each wave: 32 rows x 64 cols => 2 A-frags, 4 B-frags, 8 accumulators,
// 8 WMMAs per K-step. Async double-buffered LDS staging.
// ---------------------------------------------------------------------------
__global__ __launch_bounds__(256)
void gemm_bf16_kernel(const __bf16* __restrict__ A, const __bf16* __restrict__ W,
                      float* __restrict__ Cf, __bf16* __restrict__ Cb,
                      int M, int N, int K) {
    __shared__ __bf16 As[2][128][40];   // +8 pad, 80B rows (16B aligned)
    __shared__ __bf16 Ws[2][128][40];

    const int tid  = threadIdx.x;
    const int lane = tid & 31;
    const int wave = tid >> 5;
    const int mw   = wave >> 1;          // 0..3  (32-row strip)
    const int nw   = wave & 1;           // 0..1  (64-col strip)
    const int gm0  = blockIdx.y * 128;
    const int gn0  = blockIdx.x * 128;

    v8f acc[2][4] = {};                  // [m-sub16][n-sub16]

    // staging: each thread owns 16 contiguous bf16 (two b128) of one row
    const int srow = tid >> 1;           // 0..127
    const int sco  = (tid & 1) * 16;     // 0 or 16
    const __bf16* Ag = A + (size_t)(gm0 + srow) * K + sco;
    const __bf16* Wg = W + (size_t)(gn0 + srow) * K + sco;

    // prologue: stage K-tile 0 into buffer 0
    {
        const uint32_t la = lds_off(&As[0][srow][sco]);
        const uint32_t lw = lds_off(&Ws[0][srow][sco]);
        async_copy_b128(la,      Ag);
        async_copy_b128(la + 16, Ag + 8);
        async_copy_b128(lw,      Wg);
        async_copy_b128(lw + 16, Wg + 8);
        async_wait0();
    }

    int buf = 0;
    for (int kb = 0; kb < K; kb += 32, buf ^= 1) {
        __syncthreads();                 // tile 'buf' filled by all waves; buf^1 free
        if (kb + 32 < K) {               // prefetch next K-tile into buf^1
            const uint32_t la = lds_off(&As[buf ^ 1][srow][sco]);
            const uint32_t lw = lds_off(&Ws[buf ^ 1][srow][sco]);
            async_copy_b128(la,      Ag + kb + 32);
            async_copy_b128(la + 16, Ag + kb + 40);
            async_copy_b128(lw,      Wg + kb + 32);
            async_copy_b128(lw + 16, Wg + kb + 40);
        }

        // A fragments (rows r0, r0+16), lane halves hold K 0-15 vs 16-31
        v16bf a0, a1;
        {
            const int kh = (lane >> 4) * 16;
            const __bf16* p0 = &As[buf][mw * 32      + (lane & 15)][kh];
            const __bf16* p1 = &As[buf][mw * 32 + 16 + (lane & 15)][kh];
            #pragma unroll
            for (int i = 0; i < 8; i++) {
                a0[i] = p0[i]; a0[i + 8] = p0[i + 8];
                a1[i] = p1[i]; a1[i + 8] = p1[i + 8];
            }
        }
        // 4 B fragments (W^T 32x16 tiles), lane halves hold K 0-15 vs 16-31
        {
            const int koff = (lane >> 4) * 16;
            #pragma unroll
            for (int j = 0; j < 4; j++) {
                v16bf b;
                const __bf16* p = &Ws[buf][nw * 64 + j * 16 + (lane & 15)][koff];
                #pragma unroll
                for (int i = 0; i < 16; i++) b[i] = p[i];
                acc[0][j] = __builtin_amdgcn_wmma_f32_16x16x32_bf16(
                    false, a0, false, b, (short)0, acc[0][j], false, false);
                acc[1][j] = __builtin_amdgcn_wmma_f32_16x16x32_bf16(
                    false, a1, false, b, (short)0, acc[1][j], false, false);
            }
        }
        async_wait0();                   // my fills of buf^1 landed before barrier
    }

    // Epilogue: C/D layout => VGPR v: M=v (lanes 0-15) / M=v+8 (lanes 16-31)
    const int mloc = (lane >> 4) * 8;
    const int nloc = lane & 15;
    #pragma unroll
    for (int su = 0; su < 2; su++) {
        #pragma unroll
        for (int j = 0; j < 4; j++) {
            #pragma unroll
            for (int v = 0; v < 8; v++) {
                const int m = gm0 + mw * 32 + su * 16 + mloc + v;
                const int n = gn0 + nw * 64 + j * 16 + nloc;
                const float c = acc[su][j][v];
                if (Cf) Cf[(size_t)m * N + n] = c;
                if (Cb) Cb[(size_t)m * N + n] = (__bf16)c;
            }
        }
    }
}

// ---------------------------------------------------------------------------
// Pack [BS, D]=[B,S,H,DH] fp32 -> [B,H,S,DH] bf16 with optional per-row
// L2 normalization over DH and per-head temp scale (q only).
// ---------------------------------------------------------------------------
__global__ __launch_bounds__(128)
void pack_head_kernel(const float* __restrict__ src, const float* __restrict__ temp,
                      __bf16* __restrict__ dst, int do_norm) {
    const int bs = blockIdx.x;          // 0..BS-1
    const int h  = blockIdx.y;          // 0..H-1
    const int t  = threadIdx.x;         // 0..127
    const int b  = bs >> 11;            // S = 2048
    const int s  = bs & (Sc - 1);
    const float v = src[(size_t)bs * Dc + h * DHc + t];
    float scale = 1.0f;
    __shared__ float wsum[4];
    if (do_norm) {
        float ss = v * v;
        #pragma unroll
        for (int m = 16; m >= 1; m >>= 1) ss += __shfl_xor(ss, m, 32);
        const int lane = t & 31, wv = t >> 5;
        if (lane == 0) wsum[wv] = ss;
        __syncthreads();
        const float tot = wsum[0] + wsum[1] + wsum[2] + wsum[3];
        const float nrm = sqrtf(tot);
        scale = 1.0f / fmaxf(nrm, 1e-12f);
        if (temp) scale *= temp[h];
    }
    dst[(((size_t)(b * Hc + h) * Sc) + s) * DHc + t] = (__bf16)(v * scale);
}

// ---------------------------------------------------------------------------
// Pack V transposed: [BS, D]=[B,S,H,DH] fp32 -> [B,H,DH,S] bf16.
// Transposed layout makes P x V B-fragments contiguous in LDS rows.
// ---------------------------------------------------------------------------
__global__ __launch_bounds__(128)
void pack_vT_kernel(const float* __restrict__ src, __bf16* __restrict__ dst) {
    const int bs = blockIdx.x;          // 0..BS-1
    const int h  = blockIdx.y;          // 0..H-1
    const int t  = threadIdx.x;         // dh = 0..127
    const int b  = bs >> 11;            // S = 2048
    const int s  = bs & (Sc - 1);
    const float v = src[(size_t)bs * Dc + h * DHc + t];
    dst[(((size_t)(b * Hc + h) * DHc) + t) * Sc + s] = (__bf16)v;
}

// ---------------------------------------------------------------------------
// Flash-attention (causal, pre-normalized q/k, temp folded into q).
// grid (S/64, B*H), 128 threads = 4 waves; wave w owns q-rows q0+16w..+15.
// K in [B,H,S,DH]; V pre-transposed [B,H,DH,S] so all WMMA operand reads
// from LDS are contiguous (no scalar column gathers).
// ---------------------------------------------------------------------------
__global__ __launch_bounds__(128)
void attn_kernel(const __bf16* __restrict__ Q, const __bf16* __restrict__ Kg,
                 const __bf16* __restrict__ VgT, __bf16* __restrict__ O) {
    __shared__ __bf16 Kt[64][136];      // 64 keys x 128 dims (+pad)
    __shared__ __bf16 VtT[128][72];     // 128 dims x 64 keys (+pad)
    __shared__ __bf16 Pt[64][72];       // P tile (bf16) for A-layout restage

    const int tid  = threadIdx.x;
    const int lane = tid & 31;
    const int w    = tid >> 5;
    const int q0   = blockIdx.x * 64;
    const int bh   = blockIdx.y;
    const int b    = bh >> 4;           // H = 16
    const int h    = bh & 15;

    const __bf16* Qb  = Q   + (size_t)bh * Sc * DHc;
    const __bf16* Kb  = Kg  + (size_t)bh * Sc * DHc;
    const __bf16* VbT = VgT + (size_t)bh * DHc * Sc;

    // Q fragments for this wave's 16 rows: 4 x (16x32) over DH=128
    v16bf aq[4];
    {
        const int r  = q0 + w * 16 + (lane & 15);
        const int kh = (lane >> 4) * 16;
        const __bf16* qp = Qb + (size_t)r * DHc + kh;
        #pragma unroll
        for (int f = 0; f < 4; f++) {
            #pragma unroll
            for (int i = 0; i < 8; i++) {
                aq[f][i]     = qp[f * 32 + i];
                aq[f][i + 8] = qp[f * 32 + 8 + i];
            }
        }
    }

    v8f acc[8] = {};                    // 16 rows x 128 out dims
    float mrun[8], lrun[8];
    #pragma unroll
    for (int v = 0; v < 8; v++) { mrun[v] = -INFINITY; lrun[v] = 0.f; }

    const int mloc0 = (lane >> 4) * 8;
    const int ncol  = lane & 15;

    for (int kt = 0; kt <= q0; kt += 64) {
        __syncthreads();                // prior-iter K/V reads done before overwrite
        // Async staging: K rows (64x128) and transposed V rows (128x64)
        {
            const int r0 = tid >> 4;            // 0..7
            const int co = (tid & 15) * 8;      // bf16 col, 16B chunks
            #pragma unroll
            for (int ps = 0; ps < 8; ps++) {
                const int r = ps * 8 + r0;
                async_copy_b128(lds_off(&Kt[r][co]), Kb + (size_t)(kt + r) * DHc + co);
            }
            const int dh = tid;                 // 0..127: one V^T row per thread
            #pragma unroll
            for (int c = 0; c < 8; c++) {
                async_copy_b128(lds_off(&VtT[dh][c * 8]),
                                VbT + (size_t)dh * Sc + kt + c * 8);
            }
            async_wait0();
        }
        __syncthreads();

        // Scores: 4 x (16x16) column tiles, K-loop over DH in 32-chunks
        v8f s0 = {}, s1 = {}, s2 = {}, s3 = {};
        {
            const int koff = (lane >> 4) * 16;
            #pragma unroll
            for (int f = 0; f < 4; f++) {
                v16bf b0, b1, b2, b3;
                const __bf16* p0 = &Kt[ 0 + (lane & 15)][f * 32 + koff];
                const __bf16* p1 = &Kt[16 + (lane & 15)][f * 32 + koff];
                const __bf16* p2 = &Kt[32 + (lane & 15)][f * 32 + koff];
                const __bf16* p3 = &Kt[48 + (lane & 15)][f * 32 + koff];
                #pragma unroll
                for (int i = 0; i < 16; i++) {
                    b0[i] = p0[i]; b1[i] = p1[i]; b2[i] = p2[i]; b3[i] = p3[i];
                }
                s0 = __builtin_amdgcn_wmma_f32_16x16x32_bf16(false, aq[f], false, b0, (short)0, s0, false, false);
                s1 = __builtin_amdgcn_wmma_f32_16x16x32_bf16(false, aq[f], false, b1, (short)0, s1, false, false);
                s2 = __builtin_amdgcn_wmma_f32_16x16x32_bf16(false, aq[f], false, b2, (short)0, s2, false, false);
                s3 = __builtin_amdgcn_wmma_f32_16x16x32_bf16(false, aq[f], false, b3, (short)0, s3, false, false);
            }
        }

        // Causal mask + online softmax, per row (rows live in 16-lane halves)
        #pragma unroll
        for (int v = 0; v < 8; v++) {
            const int qrow = q0 + w * 16 + mloc0 + v;
            float sv[4];
            sv[0] = (kt +  0 + ncol > qrow) ? -INFINITY : s0[v];
            sv[1] = (kt + 16 + ncol > qrow) ? -INFINITY : s1[v];
            sv[2] = (kt + 32 + ncol > qrow) ? -INFINITY : s2[v];
            sv[3] = (kt + 48 + ncol > qrow) ? -INFINITY : s3[v];
            float rowmax = fmaxf(fmaxf(sv[0], sv[1]), fmaxf(sv[2], sv[3]));
            #pragma unroll
            for (int m = 8; m >= 1; m >>= 1)
                rowmax = fmaxf(rowmax, __shfl_xor(rowmax, m, 32));
            const float mnew = fmaxf(mrun[v], rowmax);
            const float corr = __expf(mrun[v] - mnew);
            float rsum = 0.f;
            #pragma unroll
            for (int j = 0; j < 4; j++) {
                const float p = __expf(sv[j] - mnew);
                rsum += p;
                Pt[w * 16 + mloc0 + v][j * 16 + ncol] = (__bf16)p;
            }
            #pragma unroll
            for (int m = 8; m >= 1; m >>= 1) rsum += __shfl_xor(rsum, m, 32);
            lrun[v] = lrun[v] * corr + rsum;
            mrun[v] = mnew;
            #pragma unroll
            for (int j2 = 0; j2 < 8; j2++) acc[j2][v] *= corr;
        }
        __syncthreads();                // Pt writes visible

        // P x V : A = P (16x64, 2 K-chunks), B rows from transposed V tile
        v16bf ap[2];
        {
            const int r  = w * 16 + (lane & 15);
            const int kh = (lane >> 4) * 16;
            #pragma unroll
            for (int kf = 0; kf < 2; kf++) {
                const __bf16* p = &Pt[r][kf * 32 + kh];
                #pragma unroll
                for (int i = 0; i < 8; i++) { ap[kf][i] = p[i]; ap[kf][i + 8] = p[i + 8]; }
            }
        }
        {
            const int koff = (lane >> 4) * 16;
            #pragma unroll
            for (int j2 = 0; j2 < 8; j2++) {
                #pragma unroll
                for (int kf = 0; kf < 2; kf++) {
                    v16bf bv;
                    const __bf16* p = &VtT[j2 * 16 + (lane & 15)][kf * 32 + koff];
                    #pragma unroll
                    for (int i = 0; i < 16; i++) bv[i] = p[i];
                    acc[j2] = __builtin_amdgcn_wmma_f32_16x16x32_bf16(
                        false, ap[kf], false, bv, (short)0, acc[j2], false, false);
                }
            }
        }
    }

    // Epilogue: out = acc / l, written as bf16 into [BS, D] (col = h*128+dh)
    #pragma unroll
    for (int v = 0; v < 8; v++) {
        const int r = q0 + w * 16 + mloc0 + v;
        const float inv = 1.0f / lrun[v];
        const size_t rowbase = ((size_t)b * Sc + r) * Dc + h * DHc;
        #pragma unroll
        for (int j2 = 0; j2 < 8; j2++)
            O[rowbase + j2 * 16 + ncol] = (__bf16)(acc[j2][v] * inv);
    }
}

// ---------------------------------------------------------------------------
// Gated residual: out = g*a + (1-g)*r,  g = sigmoid(go + bias[col])
// ---------------------------------------------------------------------------
__global__ void gated_residual_kernel(const float* __restrict__ go,
                                      const float* __restrict__ bias,
                                      const float* __restrict__ a,
                                      const float* __restrict__ r,
                                      float* __restrict__ out,
                                      size_t n, int ncols) {
    size_t i = (size_t)blockIdx.x * blockDim.x + threadIdx.x;
    size_t stride = (size_t)gridDim.x * blockDim.x;
    for (; i < n; i += stride) {
        const int col = (int)(i % (size_t)ncols);
        const float g = 1.f / (1.f + __expf(-(go[i] + bias[col])));
        out[i] = g * a[i] + (1.f - g) * r[i];
    }
}

// ---------------------------------------------------------------------------
// SwiGLU product: gate <- silu(gate) * up * sigmoid(gate2)   (bf16 in-place)
// ---------------------------------------------------------------------------
__global__ void ffn_prod_kernel(__bf16* __restrict__ gate,
                                const __bf16* __restrict__ up,
                                const __bf16* __restrict__ g2, size_t n) {
    size_t i = (size_t)blockIdx.x * blockDim.x + threadIdx.x;
    size_t stride = (size_t)gridDim.x * blockDim.x;
    for (; i < n; i += stride) {
        const float g  = (float)gate[i];
        const float si = g / (1.f + __expf(-g));
        const float u  = (float)up[i];
        const float q  = 1.f / (1.f + __expf(-(float)g2[i]));
        gate[i] = (__bf16)(si * u * q);
    }
}

// ---------------------------------------------------------------------------
// Host orchestration
// ---------------------------------------------------------------------------
static void launch_cast(const float* in, __bf16* out, size_t n, hipStream_t s) {
    cast_f32_bf16_kernel<<<4096, 256, 0, s>>>(in, out, n);
}
static void launch_gemm(const __bf16* A, const __bf16* W, float* Cf, __bf16* Cb,
                        int M, int N, int K, hipStream_t s) {
    dim3 g(N / 128, M / 128);
    gemm_bf16_kernel<<<g, 256, 0, s>>>(A, W, Cf, Cb, M, N, K);
}

extern "C" void kernel_launch(void* const* d_in, const int* in_sizes, int n_in,
                              void* d_out, int out_size, void* d_ws, size_t ws_size,
                              hipStream_t stream) {
    (void)in_sizes; (void)n_in; (void)out_size; (void)ws_size;

    const float* x      = (const float*)d_in[0];
    const float* q_w    = (const float*)d_in[1];
    const float* k_w    = (const float*)d_in[2];
    const float* v_w    = (const float*)d_in[3];
    const float* o_w    = (const float*)d_in[4];
    const float* temp   = (const float*)d_in[5];
    const float* ln1_w  = (const float*)d_in[6];
    const float* ln2_w  = (const float*)d_in[7];
    const float* gate_w = (const float*)d_in[8];
    const float* up_w   = (const float*)d_in[9];
    const float* gate2_w= (const float*)d_in[10];
    const float* down_w = (const float*)d_in[11];
    const float* ag_w   = (const float*)d_in[12];
    const float* ag_b   = (const float*)d_in[13];
    const float* fg_w   = (const float*)d_in[14];
    const float* fg_b   = (const float*)d_in[15];
    float* out = (float*)d_out;

    const size_t DD  = (size_t)Dc * Dc;
    const size_t DHF = (size_t)Dc * HFFc;
    const size_t BSD = (size_t)BSc * Dc;
    const size_t BSH = (size_t)BSc * HFFc;

    size_t cur = 0;
    auto alloc = [&](size_t bytes) -> void* {
        cur = (cur + 255) & ~(size_t)255;
        void* p = (char*)d_ws + cur;
        cur += bytes;
        return p;
    };

    // bf16 weight copies
    __bf16* wq   = (__bf16*)alloc(DD * 2);
    __bf16* wk   = (__bf16*)alloc(DD * 2);
    __bf16* wv   = (__bf16*)alloc(DD * 2);
    __bf16* wo   = (__bf16*)alloc(DD * 2);
    __bf16* wag  = (__bf16*)alloc(DD * 2);
    __bf16* wfg  = (__bf16*)alloc(DD * 2);
    __bf16* wgt  = (__bf16*)alloc(DHF * 2);
    __bf16* wup  = (__bf16*)alloc(DHF * 2);
    __bf16* wgt2 = (__bf16*)alloc(DHF * 2);
    __bf16* wdn  = (__bf16*)alloc(DHF * 2);
    // activations
    __bf16* hbf     = (__bf16*)alloc(BSD * 2);   // rmsnorm output (ln1 then ln2)
    __bf16* qh      = (__bf16*)alloc(BSD * 2);   // [B,H,S,DH]
    __bf16* kh      = (__bf16*)alloc(BSD * 2);   // [B,H,S,DH]
    __bf16* vhT     = (__bf16*)alloc(BSD * 2);   // [B,H,DH,S]  (transposed V)
    __bf16* attnout = (__bf16*)alloc(BSD * 2);   // attn out bf16; reused as ffnout
    __bf16* obf     = (__bf16*)alloc(BSD * 2);   // bf16 copy of o-proj output
    __bf16* gate_b  = (__bf16*)alloc(BSH * 2);   // also holds SwiGLU product
    __bf16* up_b    = (__bf16*)alloc(BSH * 2);
    __bf16* gate2_b = (__bf16*)alloc(BSH * 2);
    float*  tmp_f   = (float*)alloc(BSD * 4);    // qkv staging / ag_out / fg_out
    float*  buf_a   = (float*)alloc(BSD * 4);    // o_out (attn_out f32), then ffn_out
    float*  x1      = (float*)alloc(BSD * 4);    // after first gated residual

    // 1) weights -> bf16
    launch_cast(q_w,    wq,   DD,  stream);
    launch_cast(k_w,    wk,   DD,  stream);
    launch_cast(v_w,    wv,   DD,  stream);
    launch_cast(o_w,    wo,   DD,  stream);
    launch_cast(ag_w,   wag,  DD,  stream);
    launch_cast(fg_w,   wfg,  DD,  stream);
    launch_cast(gate_w, wgt,  DHF, stream);
    launch_cast(up_w,   wup,  DHF, stream);
    launch_cast(gate2_w,wgt2, DHF, stream);
    launch_cast(down_w, wdn,  DHF, stream);

    // 2) RMSNorm 1
    rmsnorm_kernel<<<BSc, 256, 0, stream>>>(x, ln1_w, hbf, Dc);

    // 3) q/k/v projections + l2norm/pack (temp folded into q; V transposed)
    dim3 pg(BSc, Hc);
    launch_gemm(hbf, wq, tmp_f, nullptr, BSc, Dc, Dc, stream);
    pack_head_kernel<<<pg, 128, 0, stream>>>(tmp_f, temp,    qh, 1);
    launch_gemm(hbf, wk, tmp_f, nullptr, BSc, Dc, Dc, stream);
    pack_head_kernel<<<pg, 128, 0, stream>>>(tmp_f, nullptr, kh, 1);
    launch_gemm(hbf, wv, tmp_f, nullptr, BSc, Dc, Dc, stream);
    pack_vT_kernel<<<pg, 128, 0, stream>>>(tmp_f, vhT);

    // 4) flash attention -> bf16 [BS, D]
    dim3 ag_(Sc / 64, Bc * Hc);
    attn_kernel<<<ag_, 128, 0, stream>>>(qh, kh, vhT, attnout);

    // 5) output projection (f32 + bf16), attn gate, gated residual
    launch_gemm(attnout, wo, buf_a, obf, BSc, Dc, Dc, stream);
    launch_gemm(obf, wag, tmp_f, nullptr, BSc, Dc, Dc, stream);
    gated_residual_kernel<<<4096, 256, 0, stream>>>(tmp_f, ag_b, buf_a, x, x1, BSD, Dc);

    // 6) RMSNorm 2 + SwiGLU
    rmsnorm_kernel<<<BSc, 256, 0, stream>>>(x1, ln2_w, hbf, Dc);
    launch_gemm(hbf, wgt,  nullptr, gate_b,  BSc, HFFc, Dc, stream);
    launch_gemm(hbf, wup,  nullptr, up_b,    BSc, HFFc, Dc, stream);
    launch_gemm(hbf, wgt2, nullptr, gate2_b, BSc, HFFc, Dc, stream);
    ffn_prod_kernel<<<8192, 256, 0, stream>>>(gate_b, up_b, gate2_b, BSH);

    // 7) down projection (f32 + bf16), ffn gate, final gated residual
    launch_gemm(gate_b, wdn, buf_a, attnout /*ffnout bf16*/, BSc, Dc, HFFc, stream);
    launch_gemm(attnout, wfg, tmp_f, nullptr, BSc, Dc, Dc, stream);
    gated_residual_kernel<<<4096, 256, 0, stream>>>(tmp_f, fg_b, buf_a, x1, out, BSD, Dc);
}
